// NextSequencePredictor_v3_32461362823773
// MI455X (gfx1250) — compile-verified, matrix operand order checked by reading, not run on hardware
//
#include <hip/hip_runtime.h>
#include <hip/hip_bf16.h>
#include <math.h>

// ---------------------------------------------------------------------------
// Types
// ---------------------------------------------------------------------------
typedef __attribute__((ext_vector_type(16))) __bf16 v16bf;
typedef __attribute__((ext_vector_type(8)))  float  v8f;
typedef __attribute__((ext_vector_type(4)))  unsigned int u32x4;
typedef __attribute__((ext_vector_type(8)))  int i32x8;
typedef __attribute__((ext_vector_type(4)))  int i32x4;

#define BN_SCALE 0.99950037f   // 1/sqrt(1 + 1e-3)

#if __has_builtin(__builtin_amdgcn_tensor_load_to_lds) && \
    __has_builtin(__builtin_amdgcn_s_wait_tensorcnt)
#define USE_TDM 1
#else
#define USE_TDM 0
#endif

__device__ __forceinline__ unsigned short f32_to_bf16_bits(float f) {
  union { float f; unsigned u; } v; v.f = f;
  unsigned r = v.u + 0x7FFFu + ((v.u >> 16) & 1u);   // round-to-nearest-even
  return (unsigned short)(r >> 16);
}

__device__ __forceinline__ float hsigf(float x) {
  return fminf(fmaxf(0.2f * x + 0.5f, 0.0f), 1.0f);  // Keras hard_sigmoid
}

// ---------------------------------------------------------------------------
// GEMM: C[M,N] = A * B^T. A: [Mp,Kp] bf16 (Mp=ceil16(M), zero pad),
// B: [Np,Kp] bf16 (Np=ceil64(N), zero pad). Block = 4 waves, one 16-row A
// panel per block, one 16-col B panel per wave.
// TDM path: A K-chunks (16x64B) are DMA'd into LDS double buffers by the
// Tensor Data Mover (TENSORcnt), shared by all 4 waves via ds_load; DMA of
// chunk c+1 overlaps WMMA on chunk c. Fallback: direct global b128 loads.
// ---------------------------------------------------------------------------
__global__ void k_gemm_bf16_wmma(const unsigned short* __restrict__ A,
                                 const unsigned short* __restrict__ B,
                                 float* __restrict__ C,
                                 int M, int N, int Kp) {
  const int lane  = threadIdx.x & 31;
  const int wave  = threadIdx.x >> 5;
  const int tm    = blockIdx.x;
  const int tn    = blockIdx.y * 4 + wave;
  const int ncol  = tn * 16 + (lane & 15);
  const int khalf = (lane >> 4) << 3;          // 0 for lanes 0-15, 8 for 16-31

  const char* bptr = (const char*)(B + (size_t)ncol * Kp + khalf);
  v8f acc = {};
  union Frag { v16bf v; uint4 q[2]; };

#if USE_TDM
  __shared__ __align__(16) unsigned short sA[2][16 * 32];   // 2 x 1KB chunks
  const unsigned lds0 = (unsigned)(size_t)(void*)&sA[0][0]; // LDS byte offset
  const unsigned long long ga0 =
      (unsigned long long)(size_t)(const void*)(A + (size_t)tm * 16 * Kp);
  const int nchunk = Kp / 32;

  auto issue_tdm = [&](int chunk, int buf) {
    unsigned long long ga = ga0 + (unsigned long long)chunk * 64ull;
    unsigned lds = lds0 + (unsigned)buf * (16 * 64);
    unsigned td0 = (unsigned)Kp;          // tensor_dim0 (elements of 2B)
    unsigned td1 = 0x7FFFFFFFu;           // tensor_dim1: effectively unbounded
    unsigned long long st0 = (unsigned long long)(unsigned)Kp; // dim0 stride
    u32x4 g0;
    g0[0] = 1u;                                           // count=1
    g0[1] = lds;                                          // lds_addr
    g0[2] = (unsigned)(ga & 0xFFFFFFFFull);               // global_addr lo
    g0[3] = (unsigned)((ga >> 32) & 0x01FFFFFFull) | (2u << 30); // hi + type=2
    i32x8 g1;
    g1[0] = (int)(1u << 16);                              // wg_mask=0, data_size=1 (2B)
    g1[1] = (int)((td0 & 0xFFFFu) << 16);                 // tensor_dim0 lo16
    g1[2] = (int)((td0 >> 16) | ((td1 & 0xFFFFu) << 16)); // td0 hi / td1 lo
    g1[3] = (int)((td1 >> 16) | (32u << 16));             // td1 hi / tile_dim0=32
    g1[4] = (int)(16u);                                   // tile_dim1=16, tile_dim2=0
    g1[5] = (int)(unsigned)(st0 & 0xFFFFFFFFull);         // stride0 lo32
    g1[6] = (int)(unsigned)((st0 >> 32) & 0xFFFFull);     // stride0 hi16
    g1[7] = 0;
    i32x4 gz4 = {0, 0, 0, 0};
    i32x8 gz8 = {0, 0, 0, 0, 0, 0, 0, 0};
    // 6-arg form (probe-confirmed): (g0, g1, g2, g3, g4, cpol)
    __builtin_amdgcn_tensor_load_to_lds(g0, g1, gz4, gz4, gz8, 0);
  };

  if (wave == 0) issue_tdm(0, 0);

  for (int c = 0; c < nchunk; ++c) {
    int buf = c & 1;
    if (wave == 0) __builtin_amdgcn_s_wait_tensorcnt(0);
    __syncthreads();                       // A chunk visible to all waves
    if (wave == 0 && (c + 1) < nchunk) issue_tdm(c + 1, buf ^ 1);

    Frag a, b;
    const unsigned short* arow = &sA[buf][(lane & 15) * 32 + khalf];
    a.q[0] = *(const uint4*)(arow);        // ds_load_b128
    a.q[1] = *(const uint4*)(arow + 16);   // ds_load_b128
    b.q[0] = *(const uint4*)(bptr);
    b.q[1] = *(const uint4*)(bptr + 32);
    if ((c + 1) < nchunk)
      __builtin_prefetch((const void*)(bptr + 64), 0, 3);
    bptr += 64;
    acc = __builtin_amdgcn_wmma_f32_16x16x32_bf16(
        false, a.v, false, b.v, (short)0, acc, false, false);
    __syncthreads();                       // all reads done before buf reuse
  }
#else
  const int mrow = tm * 16 + (lane & 15);
  const char* aptr = (const char*)(A + (size_t)mrow * Kp + khalf);
  for (int k = 0; k < Kp; k += 32) {
    Frag a, b;
    a.q[0] = *(const uint4*)(aptr);
    a.q[1] = *(const uint4*)(aptr + 32);
    b.q[0] = *(const uint4*)(bptr);
    b.q[1] = *(const uint4*)(bptr + 32);
    if ((k + 32) < Kp) {
      __builtin_prefetch((const void*)(aptr + 64), 0, 3);
      __builtin_prefetch((const void*)(bptr + 64), 0, 3);
    }
    aptr += 64;
    bptr += 64;
    acc = __builtin_amdgcn_wmma_f32_16x16x32_bf16(
        false, a.v, false, b.v, (short)0, acc, false, false);
  }
#endif

  // C layout: VGPR r, lanes 0-15 -> M=r, lanes 16-31 -> M=8+r ; N = lane&15
  const int col = tn * 16 + (lane & 15);
  const int rb  = tm * 16 + khalf;
  if (col < N) {
#pragma unroll
    for (int r = 0; r < 8; ++r) {
      int row = rb + r;
      if (row < M) C[(size_t)row * N + col] = acc[r];
    }
  }
}

// ---------------------------------------------------------------------------
// im2col for 3x3 conv (stride 1 or 2, TF 'SAME'), fp32 -> bf16, K padded
// ---------------------------------------------------------------------------
__global__ void k_im2col3x3(const float* __restrict__ x, unsigned short* __restrict__ out,
                            int NImg, int H, int W, int Cin,
                            int OH, int OW, int stride, int padlo, int Kp) {
  size_t idx = (size_t)blockIdx.x * blockDim.x + threadIdx.x;
  size_t total = (size_t)NImg * OH * OW * Kp;
  if (idx >= total) return;
  int k = (int)(idx % Kp);
  size_t r = idx / Kp;
  int ox = (int)(r % OW); r /= OW;
  int oy = (int)(r % OH);
  int n  = (int)(r / OH);
  unsigned short v = 0;
  if (k < 9 * Cin) {
    int ci = k % Cin; int kk = k / Cin;
    int kx = kk % 3, ky = kk / 3;
    int iy = oy * stride + ky - padlo;
    int ix = ox * stride + kx - padlo;
    if (iy >= 0 && iy < H && ix >= 0 && ix < W)
      v = f32_to_bf16_bits(x[(((size_t)n * H + iy) * W + ix) * Cin + ci]);
  }
  out[idx] = v;
}

__global__ void k_pack_conv_w(const float* __restrict__ w, unsigned short* __restrict__ out,
                              int Cin, int Cout, int Kp, int Np) {
  size_t idx = (size_t)blockIdx.x * blockDim.x + threadIdx.x;
  size_t total = (size_t)Np * Kp;
  if (idx >= total) return;
  int k  = (int)(idx % Kp);
  int co = (int)(idx / Kp);
  unsigned short v = 0;
  if (co < Cout && k < 9 * Cin) v = f32_to_bf16_bits(w[(size_t)k * Cout + co]);
  out[idx] = v;
}

__global__ void k_pack_dense_w(const float* __restrict__ w, unsigned short* __restrict__ out,
                               int K, int N, int Kp, int Np) {
  size_t idx = (size_t)blockIdx.x * blockDim.x + threadIdx.x;
  size_t total = (size_t)Np * Kp;
  if (idx >= total) return;
  int k = (int)(idx % Kp);
  int n = (int)(idx / Kp);
  unsigned short v = 0;
  if (n < N && k < K) v = f32_to_bf16_bits(w[(size_t)k * N + n]);
  out[idx] = v;
}

__global__ void k_pack_rows(const float* __restrict__ x, unsigned short* __restrict__ out,
                            int M, int K, int Kp, int Mp) {
  size_t idx = (size_t)blockIdx.x * blockDim.x + threadIdx.x;
  size_t total = (size_t)Mp * Kp;
  if (idx >= total) return;
  int k = (int)(idx % Kp);
  int m = (int)(idx / Kp);
  unsigned short v = 0;
  if (m < M && k < K) v = f32_to_bf16_bits(x[(size_t)m * K + k]);
  out[idx] = v;
}

__global__ void k_pack_concat(const float* __restrict__ x, const float* __restrict__ fan,
                              unsigned short* __restrict__ out,
                              int M, int K0, int Kf, int Kp, int Mp) {
  size_t idx = (size_t)blockIdx.x * blockDim.x + threadIdx.x;
  size_t total = (size_t)Mp * Kp;
  if (idx >= total) return;
  int k = (int)(idx % Kp);
  int m = (int)(idx / Kp);
  unsigned short v = 0;
  if (m < M) {
    if (k < K0)           v = f32_to_bf16_bits(x[(size_t)m * K0 + k]);
    else if (k < K0 + Kf) v = f32_to_bf16_bits(fan[(size_t)m * Kf + (k - K0)]);
  }
  out[idx] = v;
}

__global__ void k_bias_relu_bn(float* __restrict__ y, const float* __restrict__ bias,
                               long long total, int C, float scale) {
  size_t idx = (size_t)blockIdx.x * blockDim.x + threadIdx.x;
  if (idx >= (size_t)total) return;
  int c = (int)(idx % C);
  y[idx] = scale * fmaxf(y[idx] + bias[c], 0.0f);
}

__global__ void k_zero_f32(float* __restrict__ p, long long n) {
  size_t idx = (size_t)blockIdx.x * blockDim.x + threadIdx.x;
  if (idx < (size_t)n) p[idx] = 0.0f;
}

__global__ void k_conv_s2_direct(const float* __restrict__ x, const float* __restrict__ w,
                                 const float* __restrict__ bias, float* __restrict__ y,
                                 int NImg, int H, int W, int Cin, int Cout, float scale) {
  int OH = H >> 1, OW = W >> 1;
  size_t total = (size_t)NImg * OH * OW * Cout;
  size_t idx = (size_t)blockIdx.x * blockDim.x + threadIdx.x;
  if (idx >= total) return;
  int co = (int)(idx % Cout); size_t r = idx / Cout;
  int ox = (int)(r % OW); r /= OW;
  int oy = (int)(r % OH);
  int n  = (int)(r / OH);
  float acc = bias[co];
  for (int ky = 0; ky < 3; ++ky) {
    int iy = oy * 2 + ky;
    if (iy >= H) break;
    for (int kx = 0; kx < 3; ++kx) {
      int ix = ox * 2 + kx;
      if (ix >= W) continue;
      const float* xp = x + (((size_t)n * H + iy) * W + ix) * Cin;
      const float* wp = w + ((size_t)(ky * 3 + kx) * Cin) * Cout + co;
      for (int ci = 0; ci < Cin; ++ci) acc = fmaf(xp[ci], wp[(size_t)ci * Cout], acc);
    }
  }
  y[idx] = scale * fmaxf(acc, 0.0f);
}

__global__ void k_convt_s2(const float* __restrict__ x, const float* __restrict__ w,
                           const float* __restrict__ bias, float* __restrict__ y,
                           int NImg, int H, int W, int Cin, int Cout, float scale) {
  int OH = H * 2, OW = W * 2;
  size_t total = (size_t)NImg * OH * OW * Cout;
  size_t idx = (size_t)blockIdx.x * blockDim.x + threadIdx.x;
  if (idx >= total) return;
  int co = (int)(idx % Cout); size_t r = idx / Cout;
  int ox = (int)(r % OW); r /= OW;
  int oy = (int)(r % OH);
  int n  = (int)(r / OH);
  float acc = bias[co];
  for (int ky = 0; ky < 3; ++ky) {
    int iy2 = oy + 1 - ky;
    if (iy2 < 0 || (iy2 & 1)) continue;
    int iy = iy2 >> 1;
    if (iy >= H) continue;
    for (int kx = 0; kx < 3; ++kx) {
      int ix2 = ox + 1 - kx;
      if (ix2 < 0 || (ix2 & 1)) continue;
      int ix = ix2 >> 1;
      if (ix >= W) continue;
      const float* xp = x + (((size_t)n * H + iy) * W + ix) * Cin;
      const float* wp = w + ((size_t)(ky * 3 + kx) * Cin) * Cout + co;
      for (int ci = 0; ci < Cin; ++ci) acc = fmaf(xp[ci], wp[(size_t)ci * Cout], acc);
    }
  }
  y[idx] = scale * fmaxf(acc, 0.0f);
}

__global__ void k_lstm_step(const float* __restrict__ zx, const float* __restrict__ zh,
                            const float* __restrict__ bias,
                            float* __restrict__ cst, float* __restrict__ hst,
                            float* __restrict__ seqout,
                            int B, int T, int t, int HW, int C, float scale) {
  size_t total = (size_t)B * HW * C;
  size_t idx = (size_t)blockIdx.x * blockDim.x + threadIdx.x;
  if (idx >= total) return;
  int c = (int)(idx % C); size_t r = idx / C;
  int p = (int)(r % HW);
  int b = (int)(r / HW);
  size_t zxb = (((size_t)(b * T + t) * HW + p) * 4) * C;
  size_t zhb = (((size_t)b * HW + p) * 4) * C;
  float zi = zx[zxb + 0 * C + c] + zh[zhb + 0 * C + c] + bias[0 * C + c];
  float zf = zx[zxb + 1 * C + c] + zh[zhb + 1 * C + c] + bias[1 * C + c];
  float zc = zx[zxb + 2 * C + c] + zh[zhb + 2 * C + c] + bias[2 * C + c];
  float zo = zx[zxb + 3 * C + c] + zh[zhb + 3 * C + c] + bias[3 * C + c];
  float cn = hsigf(zf) * cst[idx] + hsigf(zi) * tanhf(zc);
  float hn = hsigf(zo) * tanhf(cn);
  cst[idx] = cn;
  hst[idx] = hn;
  seqout[(((size_t)(b * T + t) * HW) + p) * C + c] = scale * hn;
}

// ---------------------------------------------------------------------------
// Host orchestration
// ---------------------------------------------------------------------------
#define EW(K, TOTAL, ...) do {                                      \
    long long _t = (long long)(TOTAL);                              \
    unsigned _nb = (unsigned)((_t + 255) / 256);                    \
    K<<<_nb, 256, 0, stream>>>(__VA_ARGS__);                        \
  } while (0)

static inline dim3 gemm_grid(int M, int N) {
  return dim3((unsigned)((M + 15) / 16), (unsigned)((N + 63) / 64));
}
static inline int kpad32(int k) { return (k + 31) & ~31; }
static inline int npad64(int n) { return (n + 63) & ~63; }

extern "C" void kernel_launch(void* const* d_in, const int* in_sizes, int n_in,
                              void* d_out, int out_size, void* d_ws, size_t ws_size,
                              hipStream_t stream) {
  (void)in_sizes; (void)n_in; (void)out_size; (void)ws_size;

  const float* x_in = (const float*)d_in[0];   // [8,2,512,640,1]
  const float* fan  = (const float*)d_in[1];   // [8,3]
  float* out = (float*)d_out;                  // [8,2,512,640,1]

  char* ws = (char*)d_ws;
  size_t off = 0;
  auto alloc = [&](size_t bytes) -> void* {
    void* p = ws + off;
    off += (bytes + 255) & ~(size_t)255;
    return p;
  };
  float*          actA  = (float*)alloc((size_t)16 * 256 * 320 * 8 * 4);
  float*          actB  = (float*)alloc((size_t)16 * 256 * 320 * 8 * 4);
  unsigned short* packA = (unsigned short*)alloc((size_t)13200000 * 2);
  unsigned short* packW = (unsigned short*)alloc((size_t)5300000 * 2);
  unsigned short* packU = (unsigned short*)alloc((size_t)2400000 * 2);
  float*          zx    = (float*)alloc((size_t)1280 * 1024 * 4);
  float*          zh    = (float*)alloc((size_t)640 * 1024 * 4);
  float*          cst   = (float*)alloc((size_t)8 * 80 * 256 * 4);
  float*          hst   = (float*)alloc((size_t)8 * 80 * 256 * 4);
  float*          dt1   = (float*)alloc((size_t)8 * 100 * 4);
  float*          dt2   = (float*)alloc((size_t)8 * 10240 * 4);

  const int NImg = 16;   // B*T

  EW(k_conv_s2_direct, (size_t)NImg * 256 * 320 * 8,
     x_in, (const float*)d_in[2], (const float*)d_in[3], actA,
     NImg, 512, 640, 1, 8, BN_SCALE);
  EW(k_conv_s2_direct, (size_t)NImg * 128 * 160 * 16,
     actA, (const float*)d_in[4], (const float*)d_in[5], actB,
     NImg, 256, 320, 8, 16, BN_SCALE);

  const int encH[4]  = {128, 64, 32, 16};
  const int encW_[4] = {160, 80, 40, 20};
  const int encCi[4] = {16, 32, 64, 128};
  const int encCo[4] = {32, 64, 128, 256};
  const float* src = actB;
  float* dst = actA;
  for (int i = 0; i < 4; ++i) {
    int H = encH[i], W = encW_[i], Ci = encCi[i], Co = encCo[i];
    int OH = H / 2, OW = W / 2;
    int M = NImg * OH * OW;
    int Kp = kpad32(9 * Ci);
    int Np = npad64(Co);
    const float* wgt = (const float*)d_in[2 + 2 * (i + 2)];
    const float* bia = (const float*)d_in[3 + 2 * (i + 2)];
    EW(k_pack_conv_w, (size_t)Np * Kp, wgt, packW, Ci, Co, Kp, Np);
    EW(k_im2col3x3, (size_t)M * Kp, src, packA, NImg, H, W, Ci, OH, OW, 2, 0, Kp);
    k_gemm_bf16_wmma<<<gemm_grid(M, Co), 128, 0, stream>>>(packA, packW, dst, M, Co, Kp);
    EW(k_bias_relu_bn, (size_t)M * Co, dst, bia, (long long)M * Co, Co, BN_SCALE);
    const float* t = dst; dst = (float*)src; src = t;
  }
  const float* encSeq = src;         // [16, 8, 10, 256]
  float* othSeq = dst;

  auto run_convlstm = [&](const float* seqin, float* seqout,
                          const float* W, const float* U, const float* bias, int C) {
    const int HW = 80, B = 8, T = 2;
    int C4 = 4 * C;
    int Kp = kpad32(9 * C);
    int Np = npad64(C4);
    EW(k_pack_conv_w, (size_t)Np * Kp, W, packW, C, C4, Kp, Np);
    EW(k_pack_conv_w, (size_t)Np * Kp, U, packU, C, C4, Kp, Np);
    EW(k_im2col3x3, (size_t)(B * T * HW) * Kp, seqin, packA, B * T, 8, 10, C, 8, 10, 1, 1, Kp);
    k_gemm_bf16_wmma<<<gemm_grid(B * T * HW, C4), 128, 0, stream>>>(packA, packW, zx, B * T * HW, C4, Kp);
    EW(k_zero_f32, (long long)B * HW * C, cst, (long long)B * HW * C);
    EW(k_zero_f32, (long long)B * HW * C, hst, (long long)B * HW * C);
    for (int t = 0; t < T; ++t) {
      EW(k_im2col3x3, (size_t)(B * HW) * Kp, hst, packA, B, 8, 10, C, 8, 10, 1, 1, Kp);
      k_gemm_bf16_wmma<<<gemm_grid(B * HW, C4), 128, 0, stream>>>(packA, packU, zh, B * HW, C4, Kp);
      EW(k_lstm_step, (size_t)B * HW * C, zx, zh, bias, cst, hst, seqout,
         B, T, t, HW, C, BN_SCALE);
    }
  };

  for (int j = 0; j < 3; ++j) {
    run_convlstm(encSeq, othSeq,
                 (const float*)d_in[14 + 3 * j], (const float*)d_in[15 + 3 * j],
                 (const float*)d_in[16 + 3 * j], 256);
    const float* t = encSeq; encSeq = othSeq; othSeq = (float*)t;
  }

  {
    int K0 = 40960, Kf = 3, K = K0 + Kf, Kp = kpad32(K);   // 40992
    int Np = npad64(100);                                   // 128
    EW(k_pack_concat, (size_t)16 * Kp, encSeq, fan, packA, 8, K0, Kf, Kp, 16);
    EW(k_pack_dense_w, (size_t)Np * Kp, (const float*)d_in[23], packW, K, 100, Kp, Np);
    k_gemm_bf16_wmma<<<gemm_grid(8, 100), 128, 0, stream>>>(packA, packW, dt1, 8, 100, Kp);
    EW(k_bias_relu_bn, (size_t)8 * 100, dt1, (const float*)d_in[24], 800LL, 100, BN_SCALE);

    int Kp2 = kpad32(100);   // 128
    EW(k_pack_rows, (size_t)16 * Kp2, dt1, packA, 8, 100, Kp2, 16);
    EW(k_pack_dense_w, (size_t)10240 * Kp2, (const float*)d_in[25], packW, 100, 10240, Kp2, 10240);
    k_gemm_bf16_wmma<<<gemm_grid(8, 10240), 128, 0, stream>>>(packA, packW, dt2, 8, 10240, Kp2);
    EW(k_bias_relu_bn, (size_t)8 * 10240, dt2, (const float*)d_in[26], 81920LL, 10240, BN_SCALE);
  }

  const float* decSeq = dt2;
  float* decOut = actA;
  for (int j = 0; j < 3; ++j) {
    run_convlstm(decSeq, decOut,
                 (const float*)d_in[27 + 3 * j], (const float*)d_in[28 + 3 * j],
                 (const float*)d_in[29 + 3 * j], 64);
    decSeq = decOut;
    decOut = (decOut == actA) ? actB : actA;
  }

  const int dH[6]  = {8, 16, 32, 64, 128, 256};
  const int dW_[6] = {10, 20, 40, 80, 160, 320};
  const int dCi[6] = {64, 64, 32, 16, 8, 4};
  const int dCo[6] = {64, 32, 16, 8, 4, 1};
  const float* dsrc = decSeq;
  for (int i = 0; i < 6; ++i) {
    int H = dH[i], W = dW_[i], Ci = dCi[i], Co = dCo[i];
    float* ddst = (i == 5) ? out : ((dsrc == actA) ? actB : actA);
    EW(k_convt_s2, (size_t)NImg * (2 * H) * (2 * W) * Co,
       dsrc, (const float*)d_in[36 + 2 * i], (const float*)d_in[37 + 2 * i], ddst,
       NImg, H, W, Ci, Co, BN_SCALE);
    dsrc = ddst;
  }
}
